// EGNNet_60601988547125
// MI455X (gfx1250) — compile-verified, matrix operand order checked by reading, not run on hardware
//
#include <hip/hip_runtime.h>
#include <hip/hip_bf16.h>
#include <math.h>

// Problem constants (match reference)
#define BB   4
#define NN   1024
#define KK   128
#define DD   32
#define LLAY 3
#define E_IN 65
#define E_H  130

// Padded GEMM dims
#define KP1  96    // E_IN padded to multiple of 32 (GEMM1 K)
#define NP1  144   // E_H padded to multiple of 16 (GEMM1 N)
#define KP2  160   // E_H padded to multiple of 32 (GEMM2 K)
#define CH   128   // coors-MLP hidden

typedef _Float16 h8   __attribute__((ext_vector_type(8)));
typedef _Float16 v16h __attribute__((ext_vector_type(16)));
typedef float    v8f  __attribute__((ext_vector_type(8)));

__device__ __forceinline__ float lrelu(float x) { return x > 0.f ? x : 0.1f * x; }

// Load a 16x32 f16 WMMA operand fragment from a row-major f16 matrix.
// CDNA5 16-bit A layout: lanes 0-15 -> row, halves 0-7 = K(half*8..+7),
// halves 8-15 = K(16+half*8..+7).  B uses the identical pattern with the
// transposed ([N][K]) matrix, which is why weights are staged transposed.
__device__ __forceinline__ v16h ld_frag(const _Float16* M, int ld, int row0, int col0) {
    int lane = threadIdx.x & 31;
    int r = row0 + (lane & 15);
    int c = col0 + ((lane >> 4) << 3);
    h8 a = *(const h8*)(M + r * ld + c);
    h8 b = *(const h8*)(M + r * ld + c + 16);
    v16h o;
#pragma unroll
    for (int i = 0; i < 8; ++i) { o[i] = a[i]; o[i + 8] = b[i]; }
    return o;
}

__device__ __forceinline__ v8f wmma_f16(v16h a, v16h b, v8f c) {
    return __builtin_amdgcn_wmma_f32_16x16x32_f16(false, a, false, b, (short)0, c, false, false);
}

// ---------------------------------------------------------------------------
// Weight prep: transpose + pad + convert f32 -> f16 into [fan_out][fan_in_pad]
// ---------------------------------------------------------------------------
__global__ void egnn_prep_kernel(const float* __restrict__ e1w,  // [L][65][130]
                                 const float* __restrict__ e2w,  // [L][130][32]
                                 const float* __restrict__ c1w,  // [L][32][128]
                                 _Float16* __restrict__ W1t,     // [L][144][96]
                                 _Float16* __restrict__ W2t,     // [L][32][160]
                                 _Float16* __restrict__ C1t) {   // [L][128][32]
    int idx = blockIdx.x * blockDim.x + threadIdx.x;
    if (idx < LLAY * NP1 * KP1) {
        int l = idx / (NP1 * KP1);
        int r = (idx / KP1) % NP1;   // fan_out (0..143)
        int k = idx % KP1;           // fan_in  (0..95)
        float v = (r < E_H && k < E_IN) ? e1w[(l * E_IN + k) * E_H + r] : 0.f;
        W1t[idx] = (_Float16)v;
    }
    if (idx < LLAY * 32 * KP2) {
        int l = idx / (32 * KP2);
        int r = (idx / KP2) % 32;
        int k = idx % KP2;
        float v = (k < E_H) ? e2w[(l * E_H + k) * 32 + r] : 0.f;
        W2t[idx] = (_Float16)v;
    }
    if (idx < LLAY * CH * 32) {
        int l = idx / (CH * 32);
        int r = (idx / 32) % CH;
        int k = idx % 32;
        C1t[idx] = (_Float16)c1w[(l * 32 + k) * CH + r];
    }
}

// ---------------------------------------------------------------------------
// Embed: feats0 = lrelu(feat_ @ embed_w + embed_b)    [4096x32] @ [32x32]
// ---------------------------------------------------------------------------
__global__ void egnn_embed_kernel(const float* __restrict__ feat,
                                  const float* __restrict__ w,
                                  const float* __restrict__ bvec,
                                  float* __restrict__ out) {
    int idx = blockIdx.x * blockDim.x + threadIdx.x;   // BB*NN*DD
    int n = idx >> 5, d = idx & 31;
    float a = bvec[d];
#pragma unroll 8
    for (int q = 0; q < 32; ++q) a += feat[n * 32 + q] * w[q * 32 + d];
    out[idx] = lrelu(a);
}

// ---------------------------------------------------------------------------
// KNN: per node, all-pairs sq-dist within its graph + bitonic sort, keep 128
// ---------------------------------------------------------------------------
__global__ __launch_bounds__(256) void egnn_knn_kernel(const float* __restrict__ coors,
                                                       int* __restrict__ knnIdx,
                                                       float* __restrict__ knnDist) {
    __shared__ float sx[NN];
    __shared__ float sy[NN];
    __shared__ float sd[NN];
    __shared__ int   si[NN];
    int node = blockIdx.x;
    int b = node >> 10, i = node & (NN - 1);
    int t = threadIdx.x;
    for (int q = t; q < NN; q += 256) {
        sx[q] = coors[(b * NN + q) * 2 + 0];
        sy[q] = coors[(b * NN + q) * 2 + 1];
    }
    __syncthreads();
    float cx = sx[i], cy = sy[i];
    for (int q = t; q < NN; q += 256) {
        float dx = cx - sx[q], dy = cy - sy[q];
        sd[q] = dx * dx + dy * dy;
        si[q] = q;
    }
    __syncthreads();
    // bitonic sort ascending by distance
    for (int k = 2; k <= NN; k <<= 1) {
        for (int j = k >> 1; j > 0; j >>= 1) {
            for (int q = t; q < NN; q += 256) {
                int ixj = q ^ j;
                if (ixj > q) {
                    bool up = ((q & k) == 0);
                    float d0 = sd[q], d1 = sd[ixj];
                    if ((d0 > d1) == up) {
                        sd[q] = d1; sd[ixj] = d0;
                        int tmp = si[q]; si[q] = si[ixj]; si[ixj] = tmp;
                    }
                }
            }
            __syncthreads();
        }
    }
    if (t < KK) {
        knnIdx[node * KK + t]  = si[t];
        knnDist[node * KK + t] = sd[t];
    }
}

// ---------------------------------------------------------------------------
// Fused EGNN layer: one 128-thread (4-wave) block per node.
// GEMM1: X[128x96] @ W1t' -> H1[128x144] (+bias, lrelu)
// GEMM2: H1[128x160] @ W2t' -> m_raw[128x32] (+bias, lrelu), gate, m
// GEMM3: m[128x32] @ C1t' -> Hc[128x128] (+bias, lrelu)
// then cw dots, coordinate reduction, node MLP (VALU).
// Weight fragments are hoisted per N-tile and reused across both M-tiles.
// ---------------------------------------------------------------------------
__global__ __launch_bounds__(128) void egnn_edge_kernel(
    const float* __restrict__ featsIn, const float* __restrict__ coorsIn,
    const int* __restrict__ knnIdx, const float* __restrict__ knnDist,
    const _Float16* __restrict__ W1t, const float* __restrict__ e1b,
    const _Float16* __restrict__ W2t, const float* __restrict__ e2b,
    const float* __restrict__ gatew, const float* __restrict__ gateb,
    const _Float16* __restrict__ C1t, const float* __restrict__ c1b,
    const float* __restrict__ c2w, const float* __restrict__ c2b,
    const float* __restrict__ n1w, const float* __restrict__ n1b,
    const float* __restrict__ n2w, const float* __restrict__ n2b,
    const float* __restrict__ scale,
    float* __restrict__ featsOut, float* __restrict__ coorsOut) {

    // LDS with manual aliasing:
    //  region0 [0, 32768): sX [128][96] f16  -> later Hc [128][128] f16
    //  region1 [32768, 73728): sH1 [128][160] f16 -> later sMf[128][32] f32 + sM16[128][32] f16
    //  misc    [73728, ...): small arrays
    __shared__ __align__(16) unsigned char smem[32768 + 40960 + 4096];
    _Float16* sX   = (_Float16*)smem;                    // [128][KP1]
    _Float16* sHc  = (_Float16*)smem;                    // [128][CH]
    _Float16* sH1  = (_Float16*)(smem + 32768);          // [128][KP2]
    float*    sMf  = (float*)(smem + 32768);             // [128][32]
    _Float16* sM16 = (_Float16*)(smem + 32768 + 16384);  // [128][32]
    float* sFi   = (float*)(smem + 32768 + 40960);       // 32
    float* sMsum = sFi + 32;                             // 32
    float* srel  = sMsum + 32;                           // 128*2
    float* sredx = srel + 256;                           // 128
    float* sredy = sredx + 128;                          // 128
    float* sHn   = sredy + 128;                          // 64
    float* sCi   = sHn + 64;                             // 2

    int node = blockIdx.x;
    int b = node >> 10;
    int t = threadIdx.x;
    int wave = t >> 5;
    int lane = t & 31;
    int half = lane >> 4;
    int nlo  = lane & 15;

    // Prefetch the (shared, L2-resident) f16 weight images before first use.
    {
        const char* p1 = (const char*)W1t;   // 27648 B
        const char* p2 = (const char*)W2t;   // 10240 B
        const char* p3 = (const char*)C1t;   //  8192 B
        __builtin_prefetch(p1 + t * 128, 0, 0);          // 16 KB span
        __builtin_prefetch(p1 + 16384 + t * 128, 0, 0);  // rest of W1t
        __builtin_prefetch(p2 + t * 80, 0, 0);
        __builtin_prefetch(p3 + t * 64, 0, 0);
    }

    if (t < 2)  sCi[t] = coorsIn[node * 2 + t];
    if (t < 32) { sFi[t] = featsIn[node * 32 + t]; sMsum[t] = 0.f; }
    __syncthreads();
    float cix = sCi[0], ciy = sCi[1];

    // ---- Phase 0: build edge-input rows, rel coords, zero H1 pad cols ----
    {
        int j = knnIdx[node * KK + t];
        int g = (b << 10) + j;
        float dist = knnDist[node * KK + t];
        _Float16* xr = sX + t * KP1;
#pragma unroll
        for (int d = 0; d < 32; ++d) xr[d] = (_Float16)sFi[d];
#pragma unroll
        for (int d = 0; d < 32; ++d) xr[32 + d] = (_Float16)featsIn[g * 32 + d];
        xr[64] = (_Float16)dist;
#pragma unroll
        for (int d = 65; d < KP1; ++d) xr[d] = (_Float16)0.f;
        float rx = cix - coorsIn[g * 2 + 0];
        float ry = ciy - coorsIn[g * 2 + 1];
        float nrm = fmaxf(sqrtf(rx * rx + ry * ry), 1e-8f);
        float sc = scale[0] / nrm;
        srel[t * 2 + 0] = rx * sc;
        srel[t * 2 + 1] = ry * sc;
#pragma unroll
        for (int d = NP1; d < KP2; ++d) sH1[t * KP2 + d] = (_Float16)0.f;
    }
    __syncthreads();

    // ---- GEMM1: H1 = lrelu(X @ e1_w + e1_b) ----
    // Weight fragments hoisted per n-tile, reused for both m-tiles.
    for (int nt = 0; nt < 9; ++nt) {
        v16h w[3];
#pragma unroll
        for (int kt = 0; kt < 3; ++kt) w[kt] = ld_frag(W1t, KP1, nt * 16, kt * 32);
        int n = nt * 16 + nlo;
        float bias = (n < E_H) ? e1b[n] : 0.f;
        for (int mi = 0; mi < 2; ++mi) {
            int mt = wave * 2 + mi;
            v8f acc = {0.f, 0.f, 0.f, 0.f, 0.f, 0.f, 0.f, 0.f};
#pragma unroll
            for (int kt = 0; kt < 3; ++kt) {
                v16h a = ld_frag(sX, KP1, mt * 16, kt * 32);
                acc = wmma_f16(a, w[kt], acc);
            }
#pragma unroll
            for (int v = 0; v < 8; ++v) {
                int row = mt * 16 + v + half * 8;
                sH1[row * KP2 + n] = (_Float16)lrelu(acc[v] + bias);
            }
        }
    }
    __syncthreads();

    // ---- GEMM2: m_raw = lrelu(H1 @ e2_w + e2_b) (accumulate in regs) ----
    v8f accm[2][2];
    for (int nt = 0; nt < 2; ++nt) {
        v16h w[5];
#pragma unroll
        for (int kt = 0; kt < 5; ++kt) w[kt] = ld_frag(W2t, KP2, nt * 16, kt * 32);
        for (int mi = 0; mi < 2; ++mi) {
            int mt = wave * 2 + mi;
            v8f acc = {0.f, 0.f, 0.f, 0.f, 0.f, 0.f, 0.f, 0.f};
#pragma unroll
            for (int kt = 0; kt < 5; ++kt) {
                v16h a = ld_frag(sH1, KP2, mt * 16, kt * 32);
                acc = wmma_f16(a, w[kt], acc);
            }
            accm[mi][nt] = acc;
        }
    }
    __syncthreads();   // all waves done reading sH1 -> safe to alias as sMf/sM16
    for (int mi = 0; mi < 2; ++mi) {
        int mt = wave * 2 + mi;
        for (int nt = 0; nt < 2; ++nt) {
            int n = nt * 16 + nlo;
            float bias = e2b[n];
#pragma unroll
            for (int v = 0; v < 8; ++v) {
                int row = mt * 16 + v + half * 8;
                sMf[row * 32 + n] = lrelu(accm[mi][nt][v] + bias);
            }
        }
    }
    __syncthreads();

    // ---- Soft-edge gate: m = m_raw * sigmoid(m_raw . gate_w + gate_b) ----
    {
        float g = gateb[0];
        float rowm[32];
#pragma unroll
        for (int d = 0; d < 32; ++d) { rowm[d] = sMf[t * 32 + d]; g += rowm[d] * gatew[d]; }
        g = 1.f / (1.f + __expf(-g));
#pragma unroll
        for (int d = 0; d < 32; ++d) {
            float mv = rowm[d] * g;
            sMf[t * 32 + d]  = mv;
            sM16[t * 32 + d] = (_Float16)mv;
        }
    }
    __syncthreads();

    // ---- m_i: column sums over the 128 edges ----
    if (t < 32) {
        float s = 0.f;
        for (int r = 0; r < KK; ++r) s += sMf[r * 32 + t];
        sMsum[t] = s;
    }
    __syncthreads();

    // ---- GEMM3: Hc = lrelu(m @ c1_w + c1_b)  (overwrites sX region) ----
    for (int nt = 0; nt < 8; ++nt) {
        v16h w = ld_frag(C1t, 32, nt * 16, 0);
        int n = nt * 16 + nlo;
        float bias = c1b[n];
        for (int mi = 0; mi < 2; ++mi) {
            int mt = wave * 2 + mi;
            v16h a = ld_frag(sM16, 32, mt * 16, 0);
            v8f acc = {0.f, 0.f, 0.f, 0.f, 0.f, 0.f, 0.f, 0.f};
            acc = wmma_f16(a, w, acc);
#pragma unroll
            for (int v = 0; v < 8; ++v) {
                int row = mt * 16 + v + half * 8;
                sHc[row * CH + n] = (_Float16)lrelu(acc[v] + bias);
            }
        }
    }
    __syncthreads();

    // ---- cw per edge + weighted rel-coord partials ----
    {
        float cw = c2b[0];
        for (int j2 = 0; j2 < CH; ++j2) cw += (float)sHc[t * CH + j2] * c2w[j2];
        sredx[t] = cw * srel[t * 2 + 0];
        sredy[t] = cw * srel[t * 2 + 1];
    }
    __syncthreads();

    // ---- node MLP: feats_out = lrelu([fi|m_i]@n1+b1)@n2 + b2 + fi ----
    if (t < 64) {
        float a = n1b[t];
        for (int q = 0; q < 64; ++q) {
            float x = (q < 32) ? sFi[q] : sMsum[q - 32];
            a += x * n1w[q * 64 + t];
        }
        sHn[t] = lrelu(a);
    }
    __syncthreads();
    if (t < 32) {
        float a = n2b[t];
        for (int q = 0; q < 64; ++q) a += sHn[q] * n2w[q * 32 + t];
        featsOut[node * 32 + t] = a + sFi[t];
    }
    if (t == 0) {
        float sx = 0.f, sy = 0.f;
        for (int r = 0; r < KK; ++r) { sx += sredx[r]; sy += sredy[r]; }
        coorsOut[node * 2 + 0] = cix + sx;
        coorsOut[node * 2 + 1] = ciy + sy;
    }
}

// ---------------------------------------------------------------------------
extern "C" void kernel_launch(void* const* d_in, const int* in_sizes, int n_in,
                              void* d_out, int out_size, void* d_ws, size_t ws_size,
                              hipStream_t stream) {
    const float* feat_    = (const float*)d_in[0];
    const float* coor_    = (const float*)d_in[1];
    // d_in[2] = batch (unused; equal-size graphs)
    const float* embed_w  = (const float*)d_in[3];
    const float* embed_b  = (const float*)d_in[4];
    const float* e1_w     = (const float*)d_in[5];
    const float* e1_b     = (const float*)d_in[6];
    const float* e2_w     = (const float*)d_in[7];
    const float* e2_b     = (const float*)d_in[8];
    const float* gate_w   = (const float*)d_in[9];
    const float* gate_b   = (const float*)d_in[10];
    const float* c1_w     = (const float*)d_in[11];
    const float* c1_b     = (const float*)d_in[12];
    const float* c2_w     = (const float*)d_in[13];
    const float* c2_b     = (const float*)d_in[14];
    const float* n1_w     = (const float*)d_in[15];
    const float* n1_b     = (const float*)d_in[16];
    const float* n2_w     = (const float*)d_in[17];
    const float* n2_b     = (const float*)d_in[18];
    const float* coors_sc = (const float*)d_in[19];

    unsigned char* ws = (unsigned char*)d_ws;
    size_t off = 0;
    auto alloc = [&](size_t bytes) -> void* {
        void* p = ws + off;
        off += (bytes + 255) & ~(size_t)255;
        return p;
    };
    float*    featsA  = (float*)alloc((size_t)BB * NN * DD * 4);
    float*    featsB  = (float*)alloc((size_t)BB * NN * DD * 4);
    float*    coorsA  = (float*)alloc((size_t)BB * NN * 2 * 4);
    float*    coorsB  = (float*)alloc((size_t)BB * NN * 2 * 4);
    int*      knnIdx  = (int*)alloc((size_t)BB * NN * KK * 4);
    float*    knnDist = (float*)alloc((size_t)BB * NN * KK * 4);
    _Float16* W1t     = (_Float16*)alloc((size_t)LLAY * NP1 * KP1 * 2);
    _Float16* W2t     = (_Float16*)alloc((size_t)LLAY * 32 * KP2 * 2);
    _Float16* C1t     = (_Float16*)alloc((size_t)LLAY * CH * 32 * 2);

    int prepN = LLAY * NP1 * KP1;  // largest of the three regions
    egnn_prep_kernel<<<(prepN + 255) / 256, 256, 0, stream>>>(e1_w, e2_w, c1_w, W1t, W2t, C1t);
    egnn_embed_kernel<<<(BB * NN * DD) / 256, 256, 0, stream>>>(feat_, embed_w, embed_b, featsA);

    float* fin = featsA;
    const float* cin = coor_;
    for (int l = 0; l < LLAY; ++l) {
        egnn_knn_kernel<<<BB * NN, 256, 0, stream>>>(cin, knnIdx, knnDist);
        float* fout = (l == LLAY - 1) ? (float*)d_out : ((l == 0) ? featsB : featsA);
        float* cout = (l == 0) ? coorsA : ((l == 1) ? coorsB : coorsA);
        egnn_edge_kernel<<<BB * NN, 128, 0, stream>>>(
            fin, cin, knnIdx, knnDist,
            W1t + (size_t)l * NP1 * KP1, e1_b + l * E_H,
            W2t + (size_t)l * 32 * KP2,  e2_b + l * 32,
            gate_w + l * 32, gate_b + l,
            C1t + (size_t)l * CH * 32,   c1_b + l * CH,
            c2_w + l * CH, c2_b + l,
            n1_w + l * 64 * 64, n1_b + l * 64,
            n2_w + l * 64 * 32, n2_b + l * 32,
            coors_sc + l,
            fout, cout);
        fin = fout;
        cin = cout;
    }
}
